// Attention_Encoder_30897994728043
// MI455X (gfx1250) — matrix operand order repaired and checked by reading, not compiled, not run on hardware
//
#include <hip/hip_runtime.h>
#include <hip/hip_bf16.h>

typedef _Float16 half_t;
typedef __attribute__((ext_vector_type(16))) _Float16 v16h;
typedef __attribute__((ext_vector_type(8)))  _Float16 v8h;
typedef __attribute__((ext_vector_type(8)))  float    v8f;

#define WMMA_F16(a, b, c) \
  __builtin_amdgcn_wmma_f32_16x16x32_f16(false, (a), false, (b), (short)0, (c), false, false)

// ---------------------------------------------------------------------------
// Fragment loaders matching CDNA5 WMMA VGPR layouts (ISA 7.12.2, wave32).
// A (16x32 f16, row-major source): lane l holds row l%16; halves 0..7 are
// K = koff..koff+7, halves 8..15 are K = 16+koff..23+koff, koff=(l/16)*8.
// B (32x16 f16, column-major == transposed [N][K] source): lane l holds
// column l%16; halves 0..7 are K = koff..koff+7, halves 8..15 are
// K = 16+koff..23+koff, koff=(l/16)*8.
// C/D (16x16 f32): reg r -> row r+(l/16)*8, col l%16.
// ---------------------------------------------------------------------------
__device__ __forceinline__ v16h load_a_f16(const half_t* tile, int k0, int ld, int lane) {
  const int m = lane & 15;
  const int koff = (lane >> 4) << 3;
  const half_t* p = tile + (long)m * ld + k0 + koff;
  const v8h lo = *(const v8h*)p;
  const v8h hi = *(const v8h*)(p + 16);
  v16h r;
#pragma unroll
  for (int i = 0; i < 8; ++i) { r[i] = lo[i]; r[i + 8] = hi[i]; }
  return r;
}

__device__ __forceinline__ v16h load_a_f32(const float* tile, int k0, int ld, int lane) {
  const int m = lane & 15;
  const int koff = (lane >> 4) << 3;
  const float* p = tile + (long)m * ld + k0 + koff;
  v16h r;
#pragma unroll
  for (int i = 0; i < 8; ++i) { r[i] = (half_t)p[i]; r[i + 8] = (half_t)p[i + 16]; }
  return r;
}

__device__ __forceinline__ v16h load_b_t(const half_t* matT, int n0, int k0, int ld, int lane) {
  const int n = lane & 15;
  const int koff = (lane >> 4) << 3;
  const half_t* p = matT + (long)(n0 + n) * ld + k0 + koff;
  const v8h lo = *(const v8h*)p;
  const v8h hi = *(const v8h*)(p + 16);
  v16h r;
#pragma unroll
  for (int i = 0; i < 8; ++i) { r[i] = lo[i]; r[i + 8] = hi[i]; }
  return r;
}

__device__ __forceinline__ float sigmoidf(float x) { return 1.0f / (1.0f + __expf(-x)); }

// ---------------------------------------------------------------------------
// Kernel C: fold small weights.
//   id 0..2 : M{z,r,h}T[n][k] = (W? @ L?_top)[k][n]      (f32 compute -> f16)
//   id 3..5 : L{z,r,h}hT[n][k] = L?[64+k][n]
//   id 6    : fcwT[n][k] = fc_w[k][n]
// ---------------------------------------------------------------------------
__global__ void prep_weights_kernel(const float* __restrict__ fcw,
                                    const float* __restrict__ Wz, const float* __restrict__ Wr,
                                    const float* __restrict__ Wh,
                                    const float* __restrict__ Lz, const float* __restrict__ Lr,
                                    const float* __restrict__ Lh, half_t* __restrict__ wbuf) {
  const int id = blockIdx.x;
  const float* W = (id == 0) ? Wz : (id == 1) ? Wr : Wh;
  const float* L = (id == 0 || id == 3) ? Lz : (id == 1 || id == 4) ? Lr : Lh;
  for (int e = threadIdx.x; e < 4096; e += blockDim.x) {
    const int k = e & 63, n = e >> 6;
    float v;
    if (id < 3) {
      float s = 0.0f;
      for (int j = 0; j < 64; ++j) s += W[k * 64 + j] * L[j * 64 + n];  // L top rows
      v = s;
    } else if (id < 6) {
      v = L[(64 + k) * 64 + n];  // L bottom rows, transposed
    } else {
      v = fcw[k * 64 + n];       // fc_w transposed
    }
    wbuf[(long)id * 4096 + n * 64 + k] = (half_t)v;
  }
}

// Kernel A: adj f32 -> f16, vectorized.
__global__ void cvt_f32_f16_kernel(const float* __restrict__ in, half_t* __restrict__ outh, int n4) {
  const int i = blockIdx.x * blockDim.x + threadIdx.x;
  if (i < n4) {
    const float4 v = ((const float4*)in)[i];
    half_t* o = outh + (long)i * 4;
    o[0] = (half_t)v.x; o[1] = (half_t)v.y; o[2] = (half_t)v.z; o[3] = (half_t)v.w;
  }
}

// ---------------------------------------------------------------------------
// Kernel B: xhT[t][f][node] = relu(x @ fc_w + fc_b), stored feature-major f16
// so it can be consumed directly as WMMA B-fragments downstream.
// One wave computes a 16-row x 64-col tile. grid = T*N/16 = 12288.
// ---------------------------------------------------------------------------
__global__ void __launch_bounds__(32) fc_relu_kernel(const float* __restrict__ x,
                                                     const half_t* __restrict__ fcwT,
                                                     const float* __restrict__ fcb,
                                                     half_t* __restrict__ xhT) {
  const int lane = threadIdx.x;
  const long row0 = (long)blockIdx.x * 16;     // row in flattened [T*N, 64]
  const int t = (int)(row0 >> 12);             // 4096 rows per timestep
  const int node0 = (int)(row0 & 4095);
  const float* xtile = x + row0 * 64;

  v8f acc[4] = {};
#pragma unroll
  for (int kc = 0; kc < 2; ++kc) {
    const int k0 = kc * 32;
    const v16h a = load_a_f32(xtile, k0, 64, lane);
#pragma unroll
    for (int nt = 0; nt < 4; ++nt)
      acc[nt] = WMMA_F16(a, load_b_t(fcwT, nt * 16, k0, 64, lane), acc[nt]);
  }

  const int mC = (lane >> 4) << 3;
  const int nC = lane & 15;
#pragma unroll
  for (int nt = 0; nt < 4; ++nt) {
    const int n = nt * 16 + nC;
    const float bv = fcb[n];
#pragma unroll
    for (int r = 0; r < 8; ++r) {
      float v = acc[nt][r] + bv;
      v = v > 0.0f ? v : 0.0f;
      xhT[((long)t * 64 + n) * 4096 + node0 + mC + r] = (half_t)v;
    }
  }
}

// ---------------------------------------------------------------------------
// Kernel D: ax[t] = adj_f16 @ xh[t]  (the ~103-GFLOP bulk, fully parallel).
// grid (4096/64, 48); one wave computes a 64x64 output tile: 4 M-tiles share
// each B-fragment, giving 16 WMMAs per 16 b128-class loads per 32-K chunk
// (vs 4 WMMAs per 10 loads with 16-row tiles). K = 4096 in steps of 32,
// 2048 WMMAs per wave. adj (33MB f16) + xhT (25MB) are L2-resident (192MB).
// ---------------------------------------------------------------------------
__global__ void __launch_bounds__(32) spmm_kernel(const half_t* __restrict__ adjh,
                                                  const half_t* __restrict__ xhT,
                                                  half_t* __restrict__ ax) {
  const int lane = threadIdx.x;
  const int m0 = blockIdx.x * 64;
  const int t = blockIdx.y;
  const half_t* atile = adjh + (long)m0 * 4096;
  const half_t* btile = xhT + (long)t * 64 * 4096;

  v8f acc[4][4] = {};
  for (int k0 = 0; k0 < 4096; k0 += 32) {
    // prefetch this lane's adj rows ~8 iterations ahead (gfx1250 global_prefetch)
    __builtin_prefetch(atile + (long)(lane & 15) * 4096 + k0 + 256, 0, 1);

    v16h b[4];
#pragma unroll
    for (int nt = 0; nt < 4; ++nt) b[nt] = load_b_t(btile, nt * 16, k0, 4096, lane);

#pragma unroll
    for (int mt = 0; mt < 4; ++mt) {
      const v16h a = load_a_f16(atile + (long)mt * 16 * 4096, k0, 4096, lane);
#pragma unroll
      for (int nt = 0; nt < 4; ++nt)
        acc[mt][nt] = WMMA_F16(a, b[nt], acc[mt][nt]);
    }
  }

  half_t* o = ax + ((long)t * 4096 + m0) * 64;
  const int mC = (lane >> 4) << 3;
  const int nC = lane & 15;
#pragma unroll
  for (int mt = 0; mt < 4; ++mt)
#pragma unroll
    for (int nt = 0; nt < 4; ++nt)
#pragma unroll
      for (int r = 0; r < 8; ++r)
        o[(mt * 16 + mC + r) * 64 + nt * 16 + nC] = (half_t)acc[mt][nt][r];
}

// ---------------------------------------------------------------------------
// Kernel E: GRU scan. Row-independent recurrence -> one wave owns 16 nodes,
// keeps h (16x64 f16) in LDS, iterates all 48 steps locally. Per step:
//   z = sig(ax@Mz + h@Lzh + bz), r = sig(ax@Mr + h@Lrh + br)
//   ht = tanh(ax@Mh + (h*r)@Lhh + bh),  h = z*h + (1-z)*ht
// 48 WMMAs per step. Explicit s_wait_dscnt guards LDS RAW between the
// elementwise phases and the A-fragment ds reads.
// ---------------------------------------------------------------------------
__global__ void __launch_bounds__(32) tgcn_scan_kernel(const half_t* __restrict__ ax,
                                                       const half_t* __restrict__ wbuf,
                                                       const float* __restrict__ bz,
                                                       const float* __restrict__ br,
                                                       const float* __restrict__ bh,
                                                       float* __restrict__ out) {
  __shared__ __align__(32) half_t hbuf[16 * 64];
  __shared__ __align__(32) half_t hrbuf[16 * 64];
  const int lane = threadIdx.x;
  const int row0 = blockIdx.x * 16;

  const half_t* MzT  = wbuf + 0 * 4096;
  const half_t* MrT  = wbuf + 1 * 4096;
  const half_t* MhT  = wbuf + 2 * 4096;
  const half_t* LzhT = wbuf + 3 * 4096;
  const half_t* LrhT = wbuf + 4 * 4096;
  const half_t* LhhT = wbuf + 5 * 4096;

  for (int e = lane; e < 16 * 64; e += 32) hbuf[e] = (half_t)0.0f;
  asm volatile("s_wait_dscnt 0" ::: "memory");

  const int mC = (lane >> 4) << 3;
  const int nC = lane & 15;
  float bzv[4], brv[4], bhv[4];
#pragma unroll
  for (int nt = 0; nt < 4; ++nt) {
    bzv[nt] = bz[nt * 16 + nC];
    brv[nt] = br[nt * 16 + nC];
    bhv[nt] = bh[nt * 16 + nC];
  }

  for (int t = 0; t < 48; ++t) {
    const half_t* axt = ax + ((long)t * 4096 + row0) * 64;

    v8f zac[4] = {}, rac[4] = {};
#pragma unroll
    for (int kc = 0; kc < 2; ++kc) {
      const int k0 = kc * 32;
      const v16h a_ax = load_a_f16(axt, k0, 64, lane);
      const v16h a_h  = load_a_f16(hbuf, k0, 64, lane);
#pragma unroll
      for (int nt = 0; nt < 4; ++nt) {
        zac[nt] = WMMA_F16(a_ax, load_b_t(MzT,  nt * 16, k0, 64, lane), zac[nt]);
        zac[nt] = WMMA_F16(a_h,  load_b_t(LzhT, nt * 16, k0, 64, lane), zac[nt]);
        rac[nt] = WMMA_F16(a_ax, load_b_t(MrT,  nt * 16, k0, 64, lane), rac[nt]);
        rac[nt] = WMMA_F16(a_h,  load_b_t(LrhT, nt * 16, k0, 64, lane), rac[nt]);
      }
    }

    // hr = h * sigmoid(r) into LDS (C-layout element ownership per lane)
#pragma unroll
    for (int nt = 0; nt < 4; ++nt)
#pragma unroll
      for (int r = 0; r < 8; ++r) {
        const int m = mC + r, n = nt * 16 + nC;
        const float rv = sigmoidf(rac[nt][r] + brv[nt]);
        hrbuf[m * 64 + n] = (half_t)((float)hbuf[m * 64 + n] * rv);
      }
    asm volatile("s_wait_dscnt 0" ::: "memory");

    v8f hac[4] = {};
#pragma unroll
    for (int kc = 0; kc < 2; ++kc) {
      const int k0 = kc * 32;
      const v16h a_ax = load_a_f16(axt, k0, 64, lane);
      const v16h a_hr = load_a_f16(hrbuf, k0, 64, lane);
#pragma unroll
      for (int nt = 0; nt < 4; ++nt) {
        hac[nt] = WMMA_F16(a_ax, load_b_t(MhT,  nt * 16, k0, 64, lane), hac[nt]);
        hac[nt] = WMMA_F16(a_hr, load_b_t(LhhT, nt * 16, k0, 64, lane), hac[nt]);
      }
    }

#pragma unroll
    for (int nt = 0; nt < 4; ++nt)
#pragma unroll
      for (int r = 0; r < 8; ++r) {
        const int m = mC + r, n = nt * 16 + nC;
        const float zv = sigmoidf(zac[nt][r] + bzv[nt]);
        const float ht = tanhf(hac[nt][r] + bhv[nt]);
        const float hn = zv * (float)hbuf[m * 64 + n] + (1.0f - zv) * ht;
        hbuf[m * 64 + n] = (half_t)hn;
      }
    asm volatile("s_wait_dscnt 0" ::: "memory");
  }

  for (int e = lane; e < 16 * 64; e += 32) {
    const int m = e >> 6, n = e & 63;
    out[(long)(row0 + m) * 64 + n] = (float)hbuf[e];
  }
}

// ---------------------------------------------------------------------------
// Launch. Inputs (setup_inputs order):
// 0:x 1:adj 2:fc_w 3:fc_b 4:Wz 5:Wr 6:Wh 7:Lz 8:Lr 9:Lh 10:bz 11:br 12:bh
// Workspace (f16): xhT[48*64*4096] | adjh[4096*4096] | ax[48*4096*64] | wbuf[7*4096]
// ~84 MB total.
// ---------------------------------------------------------------------------
extern "C" void kernel_launch(void* const* d_in, const int* in_sizes, int n_in,
                              void* d_out, int out_size, void* d_ws, size_t ws_size,
                              hipStream_t stream) {
  (void)in_sizes; (void)n_in; (void)out_size; (void)ws_size;
  const float* x   = (const float*)d_in[0];
  const float* adj = (const float*)d_in[1];
  const float* fcw = (const float*)d_in[2];
  const float* fcb = (const float*)d_in[3];
  const float* Wz  = (const float*)d_in[4];
  const float* Wr  = (const float*)d_in[5];
  const float* Wh  = (const float*)d_in[6];
  const float* Lz  = (const float*)d_in[7];
  const float* Lr  = (const float*)d_in[8];
  const float* Lh  = (const float*)d_in[9];
  const float* bz  = (const float*)d_in[10];
  const float* br  = (const float*)d_in[11];
  const float* bh  = (const float*)d_in[12];

  half_t* ws   = (half_t*)d_ws;
  half_t* xhT  = ws;                                  // 12,582,912 halves
  half_t* adjh = xhT  + (long)48 * 64 * 4096;         // 16,777,216 halves
  half_t* axb  = adjh + (long)4096 * 4096;            // 12,582,912 halves
  half_t* wbuf = axb  + (long)48 * 4096 * 64;         //     28,672 halves

  prep_weights_kernel<<<7, 256, 0, stream>>>(fcw, Wz, Wr, Wh, Lz, Lr, Lh, wbuf);
  cvt_f32_f16_kernel<<<16384, 256, 0, stream>>>(adj, adjh, 4096 * 4096 / 4);
  fc_relu_kernel<<<48 * 4096 / 16, 32, 0, stream>>>(x, wbuf + 6 * 4096, fcb, xhT);
  spmm_kernel<<<dim3(4096 / 64, 48), 32, 0, stream>>>(adjh, xhT, axb);
  tgcn_scan_kernel<<<4096 / 16, 32, 0, stream>>>(axb, wbuf, bz, br, bh, (float*)d_out);
}
